// CrossModalAttention_85237920956951
// MI455X (gfx1250) — matrix operand order repaired and checked by reading, not compiled, not run on hardware
//
#include <hip/hip_runtime.h>
#include <hip/hip_bf16.h>
#include <stdint.h>

// ---- types ---------------------------------------------------------------
typedef __attribute__((ext_vector_type(16))) _Float16 v16h;
typedef __attribute__((ext_vector_type(8)))  float    v8f;
typedef __attribute__((ext_vector_type(4)))  unsigned int u32x4;
typedef __attribute__((ext_vector_type(4)))  float    f32x4;

#define WMMA_F16(a, b, c) \
  __builtin_amdgcn_wmma_f32_16x16x32_f16(false, (a), false, (b), (short)0, (c), false, false)

namespace {

constexpr int kB = 16;          // batch
constexpr int kN = 2048;        // sequence
constexpr int kD = 64;          // feature dim
constexpr int kRowsTotal = kB * kN;   // 32768 flattened rows per modality
constexpr int kRowsPerBlk = 128;      // 8 waves * 16 rows
constexpr int kKB = 32;               // keys per block iteration
constexpr int kNB = kN / kKB;         // 64 key blocks

union H16 { u32x4 q[2]; v16h h; };

// 16 contiguous f16 -> v16h (two ds/global b128 loads)
__device__ inline v16h load_h16(const _Float16* p) {
  H16 u;
  u.q[0] = ((const u32x4*)p)[0];
  u.q[1] = ((const u32x4*)p)[1];
  return u.h;
}

// A-fragment (16x32 f16) for V_WMMA_F32_16X16X32_F16, built from a row of
// contiguous K values.  Lane (row = lane&15, hlf = lane>>4)
// holds elements e -> k = c*32 + hlf*16 + e (contiguous, matches B fragments).
__device__ inline v16h load_a_h(const _Float16* rowp, int c, int hlf) {
  H16 u;
  u.q[0] = *(const u32x4*)(rowp + c * 32 + hlf * 16);
  u.q[1] = *(const u32x4*)(rowp + c * 32 + hlf * 16 + 8);
  return u.h;
}

// Same A-fragment but sourced from fp32 memory (converted to f16).
__device__ inline v16h load_a_f32(const float* rowp, int c, int hlf) {
  const int b0 = c * 32 + hlf * 16;
  f32x4 p0 = *(const f32x4*)(rowp + b0);
  f32x4 p1 = *(const f32x4*)(rowp + b0 + 4);
  f32x4 p2 = *(const f32x4*)(rowp + b0 + 8);
  f32x4 p3 = *(const f32x4*)(rowp + b0 + 12);
  v16h a;
  a[0]  = (_Float16)p0.x; a[1]  = (_Float16)p0.y; a[2]  = (_Float16)p0.z; a[3]  = (_Float16)p0.w;
  a[4]  = (_Float16)p1.x; a[5]  = (_Float16)p1.y; a[6]  = (_Float16)p1.z; a[7]  = (_Float16)p1.w;
  a[8]  = (_Float16)p2.x; a[9]  = (_Float16)p2.y; a[10] = (_Float16)p2.z; a[11] = (_Float16)p2.w;
  a[12] = (_Float16)p3.x; a[13] = (_Float16)p3.y; a[14] = (_Float16)p3.z; a[15] = (_Float16)p3.w;
  return a;
}

// Async copy of one 16-byte chunk per lane: global -> LDS, no VGPR roundtrip.
// Tracked with ASYNCcnt (pair with s_wait_asynccnt before the barrier).
__device__ inline void async_copy16(const void* gsrc, void* lds_dst) {
  unsigned lds = (unsigned)(uintptr_t)lds_dst;          // LDS byte address (low 32 bits)
  unsigned long long g = (unsigned long long)(uintptr_t)gsrc;
  asm volatile("global_load_async_to_lds_b128 %0, %1, off"
               :: "v"(lds), "v"(g) : "memory");
}

__device__ inline void wait_async0() {
  asm volatile("s_wait_asynccnt 0x0" ::: "memory");
}

} // namespace

// ---- kernel 1: projections  Q = (x W) * scale,  K = x W^T,  both f16 ------
__global__ void __launch_bounds__(256)
cma_proj_85237920956951(const float* __restrict__ x0, const float* __restrict__ x1,
                        const float* __restrict__ x2,
                        const float* __restrict__ W0, const float* __restrict__ W1,
                        const float* __restrict__ W2,
                        _Float16* __restrict__ Qh, _Float16* __restrict__ Kh) {
  __shared__ alignas(16) _Float16 sWq[kD * kD];  // W column-major: sWq[n*64+k] = W[k][n]
  __shared__ alignas(16) _Float16 sWk[kD * kD];  // W row-major:    sWk[n*64+k] = W[n][k]

  const int tid   = threadIdx.x;
  const int m     = blockIdx.x >> 8;                 // modality 0..2
  const int rbase = (blockIdx.x & 255) * kRowsPerBlk; // flattened (b,n) row base
  const float* x = (m == 0) ? x0 : (m == 1) ? x1 : x2;
  const float* W = (m == 0) ? W0 : (m == 1) ? W1 : W2;

  #pragma unroll
  for (int t = 0; t < 16; ++t) {
    int idx = tid + t * 256;          // 4096 = 64*64
    float w = W[idx];
    int k = idx >> 6, n = idx & 63;
    sWq[n * 64 + k] = (_Float16)w;
    sWk[idx]        = (_Float16)w;
  }
  __syncthreads();

  const int wave = tid >> 5, lane = tid & 31;
  const int hlf = lane >> 4, lane15 = lane & 15;
  const int tileRow = rbase + wave * 16;

  const float* xrow = x + (size_t)(tileRow + lane15) * kD;
  v16h a0 = load_a_f32(xrow, 0, hlf);
  v16h a1 = load_a_f32(xrow, 1, hlf);

  _Float16* Qm = Qh + (size_t)m * kRowsTotal * kD;
  _Float16* Km = Kh + (size_t)m * kRowsTotal * kD;

  #pragma unroll
  for (int n = 0; n < 4; ++n) {
    v8f aq, ak;
    #pragma unroll
    for (int g = 0; g < 8; ++g) { aq[g] = 0.0f; ak[g] = 0.0f; }
    const _Float16* bq = &sWq[(n * 16 + lane15) * 64];
    const _Float16* bk = &sWk[(n * 16 + lane15) * 64];
    aq = WMMA_F16(a0, load_h16(bq + 0 * 32 + hlf * 16), aq);
    aq = WMMA_F16(a1, load_h16(bq + 1 * 32 + hlf * 16), aq);
    ak = WMMA_F16(a0, load_h16(bk + 0 * 32 + hlf * 16), ak);
    ak = WMMA_F16(a1, load_h16(bk + 1 * 32 + hlf * 16), ak);
    #pragma unroll
    for (int g = 0; g < 8; ++g) {
      size_t row = (size_t)(tileRow + g + 8 * hlf);
      Qm[row * kD + n * 16 + lane15] = (_Float16)(aq[g] * 0.125f);  // scale folded in
      Km[row * kD + n * 16 + lane15] = (_Float16)ak[g];
    }
  }
}

// ---- kernel 2: flash attention over all 6 (i,j) pairs, mean over i --------
__global__ void __launch_bounds__(256)
cma_attn_85237920956951(const float* __restrict__ x0, const float* __restrict__ x1,
                        const float* __restrict__ x2,
                        const _Float16* __restrict__ Qh, const _Float16* __restrict__ Kh,
                        float* __restrict__ out) {
  __shared__ alignas(16) _Float16 sK[2][kKB * kD];      // [key][feat], double-buffered 8 KB
  __shared__ alignas(16) _Float16 sVT[2][kD * kKB];     // [feat][key], double-buffered 8 KB
  __shared__ alignas(16) _Float16 sP[8 * 16 * kKB];     // per-wave P tile          8 KB

  const int tid  = threadIdx.x;
  const int b    = blockIdx.x >> 4;
  const int tile = blockIdx.x & 15;
  const int wave = tid >> 5, lane = tid & 31;
  const int hlf = lane >> 4, lane15 = lane & 15;
  const int tileRow = tile * kRowsPerBlk + wave * 16;  // row within batch b

  const float* xs0 = x0 + (size_t)b * kN * kD;
  const float* xs1 = x1 + (size_t)b * kN * kD;
  const float* xs2 = x2 + (size_t)b * kN * kD;

  // synchronous V^T staging: fp32 -> f16 with transpose (needs VALU anyway)
  const int vky = tid & 31, vf0 = (tid >> 5) * 8;
  auto stage_vt = [&](const float* xblk, _Float16* dstVT) {
    const float* vp = xblk + (size_t)vky * kD + vf0;
    f32x4 q0 = *(const f32x4*)vp;
    f32x4 q1 = *(const f32x4*)(vp + 4);
    dstVT[(vf0 + 0) * kKB + vky] = (_Float16)q0.x;
    dstVT[(vf0 + 1) * kKB + vky] = (_Float16)q0.y;
    dstVT[(vf0 + 2) * kKB + vky] = (_Float16)q0.z;
    dstVT[(vf0 + 3) * kKB + vky] = (_Float16)q0.w;
    dstVT[(vf0 + 4) * kKB + vky] = (_Float16)q1.x;
    dstVT[(vf0 + 5) * kKB + vky] = (_Float16)q1.y;
    dstVT[(vf0 + 6) * kKB + vky] = (_Float16)q1.z;
    dstVT[(vf0 + 7) * kKB + vky] = (_Float16)q1.w;
  };

  float oacc[4][8];
  #pragma unroll
  for (int d = 0; d < 4; ++d)
    #pragma unroll
    for (int g = 0; g < 8; ++g) oacc[d][g] = 0.0f;

  for (int i = 0; i < 3; ++i) {
    const _Float16* qrow = Qh + ((size_t)(i * kB + b) * kN + tileRow + lane15) * kD;
    v16h qa0 = load_a_h(qrow, 0, hlf);
    v16h qa1 = load_a_h(qrow, 1, hlf);

    for (int jj = 0; jj < 2; ++jj) {
      const int j = (i + 1 + jj) % 3;
      const float*    xj = (j == 0) ? xs0 : (j == 1) ? xs1 : xs2;
      const _Float16* Kj = Kh + (size_t)(j * kB + b) * kN * kD;

      float mrow[8], lrow[8];
      v8f acc[4];
      #pragma unroll
      for (int g = 0; g < 8; ++g) { mrow[g] = -3.0e38f; lrow[g] = 0.0f; }
      #pragma unroll
      for (int d = 0; d < 4; ++d)
        #pragma unroll
        for (int g = 0; g < 8; ++g) acc[d][g] = 0.0f;

      // prologue: stage key-block 0 into buffer 0 (K async, V^T converted)
      async_copy16(Kj + (size_t)tid * 8, &sK[0][0] + (size_t)tid * 8);
      stage_vt(xj, sVT[0]);
      wait_async0();
      __syncthreads();

      for (int kb2 = 0; kb2 < kNB; ++kb2) {
        const int cur = kb2 & 1, nxt = cur ^ 1;

        // stage next key block while computing on the current one
        if (kb2 + 1 < kNB) {
          const _Float16* Kblk = Kj + (size_t)(kb2 + 1) * kKB * kD;
          async_copy16(Kblk + (size_t)tid * 8, &sK[nxt][0] + (size_t)tid * 8);
          stage_vt(xj + (size_t)(kb2 + 1) * kKB * kD, sVT[nxt]);
          if (kb2 + 2 < kNB)
            __builtin_prefetch(xj + (size_t)(kb2 + 2) * kKB * kD, 0, 0);
        }

        const _Float16* cK  = sK[cur];
        const _Float16* cVT = sVT[cur];

        // S = Q K^T (scale pre-folded into Q):  two 16x16 f32 tiles
        v8f s0, s1;
        #pragma unroll
        for (int g = 0; g < 8; ++g) { s0[g] = 0.0f; s1[g] = 0.0f; }
        s0 = WMMA_F16(qa0, load_h16(&cK[lane15 * kD + 0 * 32 + hlf * 16]), s0);
        s0 = WMMA_F16(qa1, load_h16(&cK[lane15 * kD + 1 * 32 + hlf * 16]), s0);
        s1 = WMMA_F16(qa0, load_h16(&cK[(16 + lane15) * kD + 0 * 32 + hlf * 16]), s1);
        s1 = WMMA_F16(qa1, load_h16(&cK[(16 + lane15) * kD + 1 * 32 + hlf * 16]), s1);

        // online softmax (rows 0-7 live in lanes 0-15, rows 8-15 in lanes 16-31)
        float p0[8], p1[8];
        #pragma unroll
        for (int g = 0; g < 8; ++g) {
          float v = fmaxf(s0[g], s1[g]);
          #pragma unroll
          for (int mk = 1; mk < 16; mk <<= 1) v = fmaxf(v, __shfl_xor(v, mk, 32));
          float mn = fmaxf(mrow[g], v);
          float al = __expf(mrow[g] - mn);
          float e0 = __expf(s0[g] - mn);
          float e1 = __expf(s1[g] - mn);
          float rs = e0 + e1;
          #pragma unroll
          for (int mk = 1; mk < 16; mk <<= 1) rs += __shfl_xor(rs, mk, 32);
          lrow[g] = lrow[g] * al + rs;
          mrow[g] = mn;
          p0[g] = e0; p1[g] = e1;
          acc[0][g] *= al; acc[1][g] *= al; acc[2][g] *= al; acc[3][g] *= al;
        }

        // P: C-layout -> A-layout via per-wave LDS scratch
        _Float16* sPw = sP + wave * (16 * kKB);
        #pragma unroll
        for (int g = 0; g < 8; ++g) {
          int pr = (g + 8 * hlf) * kKB;
          sPw[pr + lane15]      = (_Float16)p0[g];
          sPw[pr + 16 + lane15] = (_Float16)p1[g];
        }
        asm volatile("s_wait_dscnt 0x0" ::: "memory");
        v16h pa = load_a_h(sPw + lane15 * kKB, 0, hlf);

        // O += P @ V  (V^T staged feat-major so B-fragment reads are contiguous)
        #pragma unroll
        for (int d = 0; d < 4; ++d)
          acc[d] = WMMA_F16(pa, load_h16(&cVT[(d * 16 + lane15) * kKB + hlf * 16]), acc[d]);

        // my async stage of the next buffer must be in LDS before the barrier
        wait_async0();
        __syncthreads();
      }

      #pragma unroll
      for (int d = 0; d < 4; ++d)
        #pragma unroll
        for (int g = 0; g < 8; ++g) oacc[d][g] += acc[d][g] / lrow[g];
    }
  }

  const float inv3 = 1.0f / 3.0f;
  #pragma unroll
  for (int d = 0; d < 4; ++d)
    #pragma unroll
    for (int g = 0; g < 8; ++g)
      out[(size_t)(b * kN + tileRow + g + 8 * hlf) * kD + d * 16 + lane15] =
          oacc[d][g] * inv3;
}

// ---- host glue ------------------------------------------------------------
extern "C" void kernel_launch(void* const* d_in, const int* in_sizes, int n_in,
                              void* d_out, int out_size, void* d_ws, size_t ws_size,
                              hipStream_t stream) {
  const float* x0 = (const float*)d_in[0];
  const float* x1 = (const float*)d_in[1];
  const float* x2 = (const float*)d_in[2];
  const float* W0 = (const float*)d_in[3];
  const float* W1 = (const float*)d_in[4];
  const float* W2 = (const float*)d_in[5];

  _Float16* Qh = (_Float16*)d_ws;                       // 12 MB (3 x 32768 x 64 f16)
  _Float16* Kh = Qh + (size_t)3 * kRowsTotal * kD;      // 12 MB

  cma_proj_85237920956951<<<3 * (kRowsTotal / kRowsPerBlk), 256, 0, stream>>>(
      x0, x1, x2, W0, W1, W2, Qh, Kh);
  cma_attn_85237920956951<<<kB * (kN / kRowsPerBlk), 256, 0, stream>>>(
      x0, x1, x2, Qh, Kh, (float*)d_out);
}
